// SelfAttention_20126216749873
// MI455X (gfx1250) — matrix operand order repaired and checked by reading, not compiled, not run on hardware
//
#include <hip/hip_runtime.h>

// ---------------------------------------------------------------------------
// Self-attention (transposed-contraction variant) for MI455X / gfx1250.
// B=4, C=512, N=4096, d=64.  ~88 GFLOP, compute-bound -> bf16 WMMA, f32 acc.
// Pipeline:
//   k_xpose : x (B,C,N) f32 -> xbT (B,N,C) bf16        (tiled transpose)
//   k_wcat  : [Wq;Wk;Wv] -> Wcat (640,512) bf16 + bcat (640) f32
//   k_proj  : WMMA GEMM  Out(640,N) = Wcat @ x ; routes rows to
//             Qb (B,N,64), Kb (B,N,64)  (= K^T), Vb (B,512,N)   all bf16
//   k_stats : rls[b,j] = 1 / sum_m exp(q_j . k_m)      (WMMA QK^T)
//   k_attn  : out[o,m] = gamma * sum_j V[o,j]*exp(q_j.k_m)*rls[j] + x[o,m]
//             flash-style: never materializes the 268MB affinity matrix.
//             m-tile = 64 so all 8 waves produce S tiles and V L2 traffic
//             is halved vs a 32-wide tile.
// ---------------------------------------------------------------------------

#define NB 4
#define NC 512
#define NN 4096
#define ND 64
#define NO 640   // 64 q + 64 k + 512 v rows

typedef __attribute__((ext_vector_type(16))) __bf16 v16bf;
typedef __attribute__((ext_vector_type(8)))  float  v8f;

union ABOp  { v16bf v; uint4 q[2]; };
union PPack { uint4 q; unsigned short s[8]; };

__device__ __forceinline__ unsigned short f32_to_bf16(float f) {
  unsigned int u = __float_as_uint(f);
  u += 0x7FFFu + ((u >> 16) & 1u);          // round-to-nearest-even
  return (unsigned short)(u >> 16);
}

__device__ __forceinline__ v8f wmma_bf16(const ABOp& a, const ABOp& b, v8f c) {
  return __builtin_amdgcn_wmma_f32_16x16x32_bf16(false, a.v, false, b.v,
                                                 (short)0, c, false, false);
}

// --------------------------- x transpose to bf16 ---------------------------
__global__ void k_xpose(const float* __restrict__ x, unsigned short* __restrict__ xbT) {
  __shared__ float tile[64][65];
  const int b  = blockIdx.z;
  const int c0 = blockIdx.y * 64;
  const int n0 = blockIdx.x * 64;
  const int tx = threadIdx.x & 63;
  const int ty = threadIdx.x >> 6;          // 0..3
  const float* src = x + ((size_t)b * NC + c0) * NN + n0;
  for (int i = 0; i < 16; ++i) {
    int c = ty + i * 4;
    tile[c][tx] = src[(size_t)c * NN + tx];
  }
  __syncthreads();
  unsigned short* dst = xbT + ((size_t)b * NN + n0) * NC + c0;
  for (int i = 0; i < 16; ++i) {
    int n = ty + i * 4;
    dst[(size_t)n * NC + tx] = f32_to_bf16(tile[tx][n]);
  }
}

// ----------------------- weight concat / bf16 convert ----------------------
__global__ void k_wcat(const float* __restrict__ Wq, const float* __restrict__ Wk,
                       const float* __restrict__ Wv, const float* __restrict__ bq,
                       const float* __restrict__ bk, const float* __restrict__ bv,
                       unsigned short* __restrict__ Wcat, float* __restrict__ bcat) {
  int idx = blockIdx.x * 256 + threadIdx.x;          // 0 .. 640*512-1
  int o = idx >> 9, c = idx & 511;
  float w = (o < 64) ? Wq[o * NC + c]
          : (o < 128) ? Wk[(o - 64) * NC + c]
                      : Wv[(o - 128) * NC + c];
  Wcat[idx] = f32_to_bf16(w);
  if (c == 0)
    bcat[o] = (o < 64) ? bq[o] : (o < 128) ? bk[o - 64] : bv[o - 128];
}

// ------------------------------ projections --------------------------------
// 40960 output tiles (16o x 16n), one per wave, K = 512 -> 16 WMMA each.
__global__ void k_proj(const unsigned short* __restrict__ xbT,
                       const unsigned short* __restrict__ Wcat,
                       const float* __restrict__ bcat,
                       unsigned short* __restrict__ Qb,
                       unsigned short* __restrict__ Kb,
                       unsigned short* __restrict__ Vb) {
  const int lane = threadIdx.x & 31;
  const int wave = threadIdx.x >> 5;
  const int row  = lane & 15;
  const int h    = lane >> 4;
  const int tile = blockIdx.x * 8 + wave;
  const int b    = tile / (40 * 256);
  const int rem  = tile % (40 * 256);
  const int ot   = rem / 256;
  const int nt   = rem % 256;

  const unsigned short* wrow = Wcat + (size_t)(ot * 16 + row) * NC;
  const unsigned short* xrow = xbT + ((size_t)b * NN + nt * 16 + row) * NC;

  v8f acc = {};
  for (int cs = 0; cs < NC; cs += 32) {
    ABOp a, bb;
    a.q[0]  = *(const uint4*)(wrow + cs + 8 * h);        // A: row o, K run 0
    a.q[1]  = *(const uint4*)(wrow + cs + 16 + 8 * h);   // A: row o, K run 1
    bb.q[0] = *(const uint4*)(xrow + cs + 8 * h);        // B: col n, K run 0
    bb.q[1] = *(const uint4*)(xrow + cs + 16 + 8 * h);
    acc = wmma_bf16(a, bb, acc);
  }

  const int n  = nt * 16 + row;             // output column (pixel)
  const int ob = ot * 16 + 8 * h;           // first output row this lane holds
  if (ot < 8) {                             // Q or K rows: store (N, 64) layout
    unsigned short* dst = (ot < 4) ? Qb : Kb;
    int o2 = (ot < 4) ? ob : ob - 64;
    PPack p;
    for (int r = 0; r < 8; ++r) p.s[r] = f32_to_bf16(acc[r] + bcat[ob + r]);
    *(uint4*)(dst + ((size_t)b * NN + n) * ND + o2) = p.q;
  } else {                                  // V rows: store (512, N) layout
    int o2 = ob - 128;
    for (int r = 0; r < 8; ++r)
      Vb[((size_t)b * NC + o2 + r) * NN + n] = f32_to_bf16(acc[r] + bcat[ob + r]);
  }
}

// --------------------- softmax denominator (per query) ---------------------
__global__ void k_stats(const unsigned short* __restrict__ Qb,
                        const unsigned short* __restrict__ Kb,
                        float* __restrict__ rls) {
  const int lane = threadIdx.x & 31;
  const int wave = threadIdx.x >> 5;
  const int row  = lane & 15;
  const int h    = lane >> 4;
  const int wid  = blockIdx.x * 8 + wave;   // 0..1023
  const int b    = wid >> 8;
  const int jt   = wid & 255;               // 16-query tile

  const unsigned short* qrow = Qb + ((size_t)b * NN + jt * 16 + row) * ND;
  ABOp a0, a1;
  a0.q[0] = *(const uint4*)(qrow + 8 * h);
  a0.q[1] = *(const uint4*)(qrow + 16 + 8 * h);
  a1.q[0] = *(const uint4*)(qrow + 32 + 8 * h);
  a1.q[1] = *(const uint4*)(qrow + 48 + 8 * h);

  float sacc[8];
  for (int r = 0; r < 8; ++r) sacc[r] = 0.f;

  for (int m = 0; m < NN; m += 16) {
    const unsigned short* krow = Kb + ((size_t)b * NN + m + row) * ND;
    ABOp b0, b1;
    b0.q[0] = *(const uint4*)(krow + 8 * h);
    b0.q[1] = *(const uint4*)(krow + 16 + 8 * h);
    b1.q[0] = *(const uint4*)(krow + 32 + 8 * h);
    b1.q[1] = *(const uint4*)(krow + 48 + 8 * h);
    v8f s = {};
    s = wmma_bf16(a0, b0, s);
    s = wmma_bf16(a1, b1, s);
    // logits are O(1) by construction (0.02 init) -> raw exp is f32-safe
    for (int r = 0; r < 8; ++r) sacc[r] += __expf(s[r]);
  }
  // reduce the 16 lanes sharing each row (xor stays within the half-wave)
  for (int mask = 1; mask <= 8; mask <<= 1)
    for (int r = 0; r < 8; ++r) sacc[r] += __shfl_xor(sacc[r], mask, 32);
  if (row == 0) {
    int jb = jt * 16 + 8 * h;
    for (int r = 0; r < 8; ++r)
      rls[(size_t)b * NN + jb + r] = 1.0f / sacc[r];
  }
}

// --------------------- fused output GEMM (flash-style) ---------------------
// One workgroup = 64 output columns m, 8 waves; wave w owns rows [64w,64w+64).
// Loop j in blocks of 32: every wave recomputes one 16x16 S tile (jt=w>>2,
// mt=w&3), forms P=exp(S)*rls in registers, stages it in LDS column-major;
// then all waves accumulate their 4x4 grid of 16x16 output tiles with
// v_wmma_f32_16x16x32_bf16 (16 WMMA per wave per j-step, 1:16 S:VP ratio).
__global__ void k_attn(const unsigned short* __restrict__ Qb,
                       const unsigned short* __restrict__ Kb,
                       const unsigned short* __restrict__ Vb,
                       const float* __restrict__ rls,
                       const float* __restrict__ xin,
                       const float* __restrict__ gamma,
                       float* __restrict__ out) {
  __shared__ unsigned short Pbuf[64][40];   // [m][j], rows 80B (16B aligned)
  const int lane = threadIdx.x & 31;
  const int wave = threadIdx.x >> 5;
  const int row  = lane & 15;
  const int h    = lane >> 4;
  const int b    = blockIdx.x >> 6;         // 64 m-blocks per batch
  const int mb   = (blockIdx.x & 63) * 64;
  const float g  = *gamma;

  v8f acc[16];                              // [ot][mt], ot=0..3, mt=0..3
  for (int i = 0; i < 16; ++i) acc[i] = (v8f){};

  const int jt = wave >> 2;                 // S-tile assignment: jt 0..1
  const int mt = wave & 3;                  //                    mt 0..3
  // K B-operands are fixed per workgroup: load once per wave
  ABOp kb0, kb1;
  {
    const unsigned short* krow = Kb + ((size_t)b * NN + mb + mt * 16 + row) * ND;
    kb0.q[0] = *(const uint4*)(krow + 8 * h);
    kb0.q[1] = *(const uint4*)(krow + 16 + 8 * h);
    kb1.q[0] = *(const uint4*)(krow + 32 + 8 * h);
    kb1.q[1] = *(const uint4*)(krow + 48 + 8 * h);
  }

  for (int jb = 0; jb < NN; jb += 32) {
    {
      const unsigned short* qrow = Qb + ((size_t)b * NN + jb + jt * 16 + row) * ND;
      ABOp qa0, qa1;
      qa0.q[0] = *(const uint4*)(qrow + 8 * h);
      qa0.q[1] = *(const uint4*)(qrow + 16 + 8 * h);
      qa1.q[0] = *(const uint4*)(qrow + 32 + 8 * h);
      qa1.q[1] = *(const uint4*)(qrow + 48 + 8 * h);
      v8f s = {};
      s = wmma_bf16(qa0, kb0, s);
      s = wmma_bf16(qa1, kb1, s);
      const float* rp = rls + (size_t)b * NN + jb + jt * 16 + 8 * h;
      float4 r0 = *(const float4*)(rp);
      float4 r1 = *(const float4*)(rp + 4);
      PPack p;
      p.s[0] = f32_to_bf16(__expf(s[0]) * r0.x);
      p.s[1] = f32_to_bf16(__expf(s[1]) * r0.y);
      p.s[2] = f32_to_bf16(__expf(s[2]) * r0.z);
      p.s[3] = f32_to_bf16(__expf(s[3]) * r0.w);
      p.s[4] = f32_to_bf16(__expf(s[4]) * r1.x);
      p.s[5] = f32_to_bf16(__expf(s[5]) * r1.y);
      p.s[6] = f32_to_bf16(__expf(s[6]) * r1.z);
      p.s[7] = f32_to_bf16(__expf(s[7]) * r1.w);
      *(uint4*)(&Pbuf[mt * 16 + row][jt * 16 + 8 * h]) = p.q;
    }
    __syncthreads();
    for (int ot = 0; ot < 4; ++ot) {
      const unsigned short* vrow =
          Vb + ((size_t)b * NC + wave * 64 + ot * 16 + row) * NN;
      ABOp va;
      va.q[0] = *(const uint4*)(vrow + jb + 8 * h);
      va.q[1] = *(const uint4*)(vrow + jb + 16 + 8 * h);
      if (jb + 32 < NN)                      // pull next j-block into WGP$
        __builtin_prefetch(vrow + jb + 32 + 8 * h, 0, 3);
      for (int mtt = 0; mtt < 4; ++mtt) {
        ABOp pb;
        pb.q[0] = *(const uint4*)(&Pbuf[mtt * 16 + row][8 * h]);
        pb.q[1] = *(const uint4*)(&Pbuf[mtt * 16 + row][16 + 8 * h]);
        acc[ot * 4 + mtt] = wmma_bf16(va, pb, acc[ot * 4 + mtt]);
      }
    }
    __syncthreads();
  }

  // epilogue: out = gamma*acc + input (f32)
  for (int ot = 0; ot < 4; ++ot) {
    for (int mtt = 0; mtt < 4; ++mtt) {
      const int o0 = wave * 64 + ot * 16 + 8 * h;
      const int m  = mb + mtt * 16 + row;
      for (int r = 0; r < 8; ++r) {
        size_t idx = ((size_t)b * NC + o0 + r) * NN + m;
        out[idx] = g * acc[ot * 4 + mtt][r] + xin[idx];
      }
    }
  }
}

// ------------------------------- launcher ----------------------------------
extern "C" void kernel_launch(void* const* d_in, const int* in_sizes, int n_in,
                              void* d_out, int out_size, void* d_ws, size_t ws_size,
                              hipStream_t stream) {
  const float* x     = (const float*)d_in[0];
  const float* Wq    = (const float*)d_in[1];
  const float* bq    = (const float*)d_in[2];
  const float* Wk    = (const float*)d_in[3];
  const float* bk    = (const float*)d_in[4];
  const float* Wv    = (const float*)d_in[5];
  const float* bv    = (const float*)d_in[6];
  const float* gamma = (const float*)d_in[7];
  float* out = (float*)d_out;

  char* ws = (char*)d_ws;
  size_t off = 0;
  auto carve = [&](size_t bytes) -> void* {
    off = (off + 255) & ~(size_t)255;
    void* p = ws + off;
    off += bytes;
    return p;
  };
  unsigned short* xbT  = (unsigned short*)carve((size_t)NB * NN * NC * 2); // 16 MB
  unsigned short* Wcat = (unsigned short*)carve((size_t)NO * NC * 2);      // 640 KB
  float*          bcat = (float*)carve((size_t)NO * 4);
  unsigned short* Qb   = (unsigned short*)carve((size_t)NB * NN * ND * 2); // 2 MB
  unsigned short* Kb   = (unsigned short*)carve((size_t)NB * NN * ND * 2); // 2 MB
  unsigned short* Vb   = (unsigned short*)carve((size_t)NB * NC * NN * 2); // 16 MB
  float*          rls  = (float*)carve((size_t)NB * NN * 4);               // 64 KB
  (void)ws_size; (void)in_sizes; (void)n_in; (void)out_size;

  k_xpose<<<dim3(NN / 64, NC / 64, NB), 256, 0, stream>>>(x, xbT);
  k_wcat<<<(NO * NC) / 256, 256, 0, stream>>>(Wq, Wk, Wv, bq, bk, bv, Wcat, bcat);
  k_proj<<<(NB * 40 * 256) / 8, 256, 0, stream>>>(xbT, Wcat, bcat, Qb, Kb, Vb);
  k_stats<<<(NB * (NN / 16)) / 8, 256, 0, stream>>>(Qb, Kb, rls);
  k_attn<<<NB * (NN / 64), 256, 0, stream>>>(Qb, Kb, Vb, rls, x, gamma, out);
}
